// TreeLstm_43181601194163
// MI455X (gfx1250) — compile-verified, matrix operand order checked by reading, not run on hardware
//
#include <hip/hip_runtime.h>
#include <cstdint>
#include <cstddef>

// ---------------------------------------------------------------------------
// TreeLSTM for MI455X (gfx1250): bf16 WMMA (v_wmma_f32_16x16x32_bf16),
// fp32 cell state, fused cell + prediction epilogues, L2-resident packed
// weights. Compile-only target: wave32, WMMA fragment layouts per CDNA5 ISA.
// ---------------------------------------------------------------------------

typedef __attribute__((ext_vector_type(16))) __bf16 v16bf;
typedef __attribute__((ext_vector_type(8)))  float  v8f;

#define B_     64
#define L_     1024
#define V_     50000
#define WD_    300
#define D_     256
#define NL_    5
#define G_     1280        // 5 * D
#define KX_    320         // leaf K padded 300 -> 320 (multiple of 32)
#define NKX_   (KX_/32)    // 10 K-tiles for leaf
#define K2_    512         // tree K = [h_l | h_r]
#define NK2_   (K2_/32)    // 16 K-tiles for tree levels
#define NTILES_ (G_/16)    // 80 N-tiles of 16 columns
#define TOTPOS_ 2047       // 1024+512+...+1

union Frag { v16bf v; uint4 q[2]; };

__device__ __forceinline__ unsigned short f2bf(float f) {
  union { float f; unsigned u; } x; x.f = f;
  unsigned r = x.u + 0x7FFFu + ((x.u >> 16) & 1u);   // round-to-nearest-even
  return (unsigned short)(r >> 16);
}
__device__ __forceinline__ float sigm(float x) { return 1.0f / (1.0f + __expf(-x)); }

// K offset within a 32-wide K tile for (lane, j) per CDNA5 16-bit A layout:
// lanes 0-15 hold K {0..7, 16..23}; lanes 16-31 hold K {8..15, 24..31}.
__device__ __forceinline__ int koff(int lane, int j) {
  return (j & 7) + ((j >> 3) << 4) + ((lane >> 4) << 3);
}

// ---------------------------------------------------------------------------
// Pack Wx (300x1280, zero-padded to 320) and [Ul;Ur] (512x1280) from fp32
// row-major into bf16 per-lane fragment order: for each (ntile, ktile) a
// 32-lane x 16-value block, so the GEMM hot loop does two b128 loads/lane.
// Runs once per launch over ~1M elements: negligible, stays hot in L2.
// ---------------------------------------------------------------------------
__global__ void pack_weights(const float* __restrict__ Wx,
                             const float* __restrict__ Ul,
                             const float* __restrict__ Ur,
                             unsigned short* __restrict__ Wxp,
                             unsigned short* __restrict__ W2p) {
  const int nWx = NTILES_ * NKX_ * 32 * 16;   // 409600
  const int nW2 = NTILES_ * NK2_ * 32 * 16;   // 655360
  int idx = blockIdx.x * 256 + threadIdx.x;
  if (idx < nWx) {
    int j = idx & 15, rest = idx >> 4;
    int lane = rest & 31; rest >>= 5;
    int kt = rest % NKX_, nt = rest / NKX_;
    int k = kt * 32 + koff(lane, j);
    int n = nt * 16 + (lane & 15);
    float v = (k < WD_) ? Wx[(size_t)k * G_ + n] : 0.0f;
    Wxp[idx] = f2bf(v);
  } else if (idx < nWx + nW2) {
    int id = idx - nWx;
    int j = id & 15, rest = id >> 4;
    int lane = rest & 31; rest >>= 5;
    int kt = rest % NK2_, nt = rest / NK2_;
    int k = kt * 32 + koff(lane, j);
    int n = nt * 16 + (lane & 15);
    float v = (k < D_) ? Ul[(size_t)k * G_ + n] : Ur[(size_t)(k - D_) * G_ + n];
    W2p[id] = f2bf(v);
  }
}

// ---------------------------------------------------------------------------
// Leaf: gates = emb[word_ids] @ Wx + b ; (h,c) = cell(gates, 0, 0).
// Since c_l = c_r = 0, the fl/fr gate columns contribute nothing: compute
// only gate groups {i(0), o(3), u(4)} -> 6 WMMA tiles per wave instead of 10.
// Fused epilogue writes c (fp32), h (bf16 for next level), and the 5-wide
// prediction row straight to d_out via an LDS-staged 16x256 h tile.
// ---------------------------------------------------------------------------
__global__ __launch_bounds__(256) void leaf_kernel(
    const int* __restrict__ wid, const float* __restrict__ emb,
    const unsigned short* __restrict__ Wxp,
    const float* __restrict__ bias, const float* __restrict__ predW,
    const float* __restrict__ predb,
    unsigned short* __restrict__ hb, float* __restrict__ cb,
    float* __restrict__ out) {
  __shared__ alignas(16) unsigned short As[16][48];  // 16 rows x 32 K (padded)
  __shared__ float Hs[16][264];                      // h staging for pred head

  const int tid  = threadIdx.x;
  const int lane = tid & 31, w = tid >> 5;
  const int half = lane >> 4, lr = lane & 15;
  const int mblk = blockIdx.x;
  const int gateCol[3] = {0, 3, 4};                  // i, o, u groups

  v8f acc[3][2];
  for (int g = 0; g < 3; ++g)
    for (int t = 0; t < 2; ++t)
      for (int r = 0; r < 8; ++r) acc[g][t][r] = 0.0f;

  for (int kt = 0; kt < NKX_; ++kt) {
    __syncthreads();
    // Stage A tile: embedding gather + fp32->bf16, zero pad beyond K=300.
    for (int e = tid; e < 512; e += 256) {
      int row = e >> 5, kk = e & 31;
      int k = kt * 32 + kk;
      int token = mblk * 16 + row;
      float v = 0.0f;
      if (k < WD_) v = emb[(size_t)wid[token] * WD_ + k];
      As[row][kk] = f2bf(v);
    }
    __syncthreads();

    Frag af;
    {  // per-lane A fragment: two 16B LDS reads (K halves per CDNA5 layout)
      const uint4* rp = reinterpret_cast<const uint4*>(&As[lr][0]);
      af.q[0] = rp[half];
      af.q[1] = rp[2 + half];
    }
    for (int g = 0; g < 3; ++g) {
      for (int t = 0; t < 2; ++t) {
        int ntile = gateCol[g] * 16 + w * 2 + t;
        const uint4* bp = reinterpret_cast<const uint4*>(
            Wxp + ((size_t)(ntile * NKX_ + kt) * 32 + lane) * 16);
        Frag bf; bf.q[0] = bp[0]; bf.q[1] = bp[1];
        acc[g][t] = __builtin_amdgcn_wmma_f32_16x16x32_bf16(
            false, af.v, false, bf.v, (short)0, acc[g][t], false, false);
      }
    }
  }

  // Epilogue: cell with c_l = c_r = 0.
  for (int t = 0; t < 2; ++t) {
    int d = w * 32 + t * 16 + lr;
    float bi = bias[d], bo = bias[768 + d], bu = bias[1024 + d];
    for (int r = 0; r < 8; ++r) {
      int mrow = r + half * 8;                 // C/D layout: lanes 16-31 -> M+8
      int node = mblk * 16 + mrow;
      float iv = acc[0][t][r] + bi;
      float ov = acc[1][t][r] + bo;
      float uv = acc[2][t][r] + bu;
      float c = sigm(iv) * tanhf(uv);
      float h = sigm(ov) * tanhf(c);
      cb[(size_t)node * D_ + d] = c;
      hb[(size_t)node * D_ + d] = f2bf(h);
      Hs[mrow][d] = h;
    }
  }
  __syncthreads();
  if (tid < 16 * NL_) {                        // fused prediction head
    int row = tid / NL_, nl = tid % NL_;
    int node = mblk * 16 + row;
    int bidx = node >> 10;                     // / 1024
    int pos  = node & 1023;                    // level-0 posOff = 0
    float s = predb[nl];
    for (int k = 0; k < D_; ++k) s += Hs[row][k] * predW[k * NL_ + nl];
    out[((size_t)bidx * TOTPOS_ + pos) * NL_ + nl] = s;
  }
}

// ---------------------------------------------------------------------------
// Tree level: gates = h_l @ Ul + h_r @ Ur + b as one K=512 GEMM where
// A[m, k<256] = h[left child], A[m, k>=256] = h[right child] (bf16 rows,
// gathered into the LDS A tile). Full 5-gate cell with fp32 c_l/c_r.
// ---------------------------------------------------------------------------
__global__ __launch_bounds__(256) void level_kernel(
    const unsigned short* __restrict__ hb_prev, const float* __restrict__ cb_prev,
    const unsigned short* __restrict__ W2p,
    const float* __restrict__ bias, const float* __restrict__ predW,
    const float* __restrict__ predb,
    unsigned short* __restrict__ hb_cur, float* __restrict__ cb_cur,
    float* __restrict__ out,
    int lenLog2, int posOff) {
  __shared__ alignas(16) unsigned short As[16][48];
  __shared__ float Hs[16][264];

  const int tid  = threadIdx.x;
  const int lane = tid & 31, w = tid >> 5;
  const int half = lane >> 4, lr = lane & 15;
  const int mblk = blockIdx.x;
  const int lenM1 = (1 << lenLog2) - 1;

  v8f acc[5][2];
  for (int g = 0; g < 5; ++g)
    for (int t = 0; t < 2; ++t)
      for (int r = 0; r < 8; ++r) acc[g][t][r] = 0.0f;

  for (int kt = 0; kt < NK2_; ++kt) {
    __syncthreads();
    for (int e = tid; e < 512; e += 256) {
      int row = e >> 5, kk = e & 31;
      int k = kt * 32 + kk;
      int node = mblk * 16 + row;
      int b = node >> lenLog2, i = node & lenM1;
      int child = (b << (lenLog2 + 1)) + 2 * i + (k >> 8);  // left / right
      As[row][kk] = hb_prev[(size_t)child * D_ + (k & 255)];
    }
    __syncthreads();

    Frag af;
    {
      const uint4* rp = reinterpret_cast<const uint4*>(&As[lr][0]);
      af.q[0] = rp[half];
      af.q[1] = rp[2 + half];
    }
    for (int g = 0; g < 5; ++g) {
      for (int t = 0; t < 2; ++t) {
        int ntile = g * 16 + w * 2 + t;
        const uint4* bp = reinterpret_cast<const uint4*>(
            W2p + ((size_t)(ntile * NK2_ + kt) * 32 + lane) * 16);
        Frag bf; bf.q[0] = bp[0]; bf.q[1] = bp[1];
        acc[g][t] = __builtin_amdgcn_wmma_f32_16x16x32_bf16(
            false, af.v, false, bf.v, (short)0, acc[g][t], false, false);
      }
    }
  }

  for (int t = 0; t < 2; ++t) {
    int d = w * 32 + t * 16 + lr;
    float bi  = bias[d],       bl = bias[256 + d], br = bias[512 + d];
    float bo  = bias[768 + d], bu = bias[1024 + d];
    for (int r = 0; r < 8; ++r) {
      int mrow = r + half * 8;
      int node = mblk * 16 + mrow;
      int b = node >> lenLog2, i = node & lenM1;
      size_t cln = (size_t)((b << (lenLog2 + 1)) + 2 * i);
      float cl = cb_prev[cln * D_ + d];
      float cr = cb_prev[(cln + 1) * D_ + d];
      float iv = acc[0][t][r] + bi;
      float fl = acc[1][t][r] + bl;
      float fr = acc[2][t][r] + br;
      float ov = acc[3][t][r] + bo;
      float uv = acc[4][t][r] + bu;
      float c = sigm(iv) * tanhf(uv) + sigm(fl) * cl + sigm(fr) * cr;
      float h = sigm(ov) * tanhf(c);
      cb_cur[(size_t)node * D_ + d] = c;
      hb_cur[(size_t)node * D_ + d] = f2bf(h);
      Hs[mrow][d] = h;
    }
  }
  __syncthreads();
  if (tid < 16 * NL_) {
    int row = tid / NL_, nl = tid % NL_;
    int node = mblk * 16 + row;
    int bidx = node >> lenLog2;
    int pos  = posOff + (node & lenM1);
    float s = predb[nl];
    for (int k = 0; k < D_; ++k) s += Hs[row][k] * predW[k * NL_ + nl];
    out[((size_t)bidx * TOTPOS_ + pos) * NL_ + nl] = s;
  }
}

// ---------------------------------------------------------------------------
extern "C" void kernel_launch(void* const* d_in, const int* in_sizes, int n_in,
                              void* d_out, int out_size, void* d_ws, size_t ws_size,
                              hipStream_t stream) {
  (void)in_sizes; (void)n_in; (void)out_size; (void)ws_size;
  const int*   wid  = (const int*)  d_in[0];
  const float* emb  = (const float*)d_in[1];
  const float* Wx   = (const float*)d_in[2];
  const float* Ul   = (const float*)d_in[3];
  const float* Ur   = (const float*)d_in[4];
  const float* bias = (const float*)d_in[5];
  const float* pW   = (const float*)d_in[6];
  const float* pb   = (const float*)d_in[7];
  float* out = (float*)d_out;

  // Workspace carve-out (256B aligned): packed bf16 weights (L2-resident),
  // per-level bf16 h (next-level GEMM operand), per-level fp32 c state.
  char* ws = (char*)d_ws;
  size_t off = 0;
  auto take = [&](size_t bytes) -> char* {
    char* p = ws + off;
    off = (off + bytes + 255) & ~(size_t)255;
    return p;
  };
  unsigned short* Wxp = (unsigned short*)take((size_t)NTILES_ * NKX_ * 32 * 16 * 2);
  unsigned short* W2p = (unsigned short*)take((size_t)NTILES_ * NK2_ * 32 * 16 * 2);
  size_t nodeOff[11], tot = 0;
  for (int l = 0; l < 11; ++l) { nodeOff[l] = tot; tot += (size_t)B_ * (L_ >> l); }
  unsigned short* hb = (unsigned short*)take(tot * D_ * sizeof(unsigned short));
  float*          cb = (float*)take(tot * D_ * sizeof(float));

  {
    int nEl = NTILES_ * NKX_ * 32 * 16 + NTILES_ * NK2_ * 32 * 16;
    pack_weights<<<(nEl + 255) / 256, 256, 0, stream>>>(Wx, Ul, Ur, Wxp, W2p);
  }

  leaf_kernel<<<(B_ * L_) / 16, 256, 0, stream>>>(
      wid, emb, Wxp, bias, pW, pb,
      hb + nodeOff[0] * D_, cb + nodeOff[0] * D_, out);

  int posOff = L_;
  for (int l = 1; l < 11; ++l) {
    int len = L_ >> l;
    level_kernel<<<(B_ * len) / 16, 256, 0, stream>>>(
        hb + nodeOff[l - 1] * D_, cb + nodeOff[l - 1] * D_, W2p,
        bias, pW, pb,
        hb + nodeOff[l] * D_, cb + nodeOff[l] * D_, out,
        10 - l, posOff);
    posOff += len;
  }
}
